// FastGCNConv_55662776156291
// MI455X (gfx1250) — matrix operand order repaired and checked by reading, not compiled, not run on hardware
//
#include <hip/hip_runtime.h>

#define N_NODES 500000
#define IN_CH   256
#define OUT_CH  256
#define S_SAMP  100000
#define TM      32          // sampled rows per GEMM block

typedef __attribute__((ext_vector_type(16))) __bf16    v16bf;
typedef __attribute__((ext_vector_type(8)))  float     v8f;
typedef __attribute__((ext_vector_type(4)))  unsigned  uv4;

union Frag { v16bf v; uv4 q[2]; };

static __device__ __forceinline__ unsigned monokey(float f) {
  unsigned b = __float_as_uint(f);
  return (b & 0x80000000u) ? ~b : (b | 0x80000000u);
}

// Split fp32 into bf16 hi (truncate) + bf16 lo (RNE of residual).
static __device__ __forceinline__ void split_bf16(float f, unsigned short* hi, unsigned short* lo) {
  unsigned b  = __float_as_uint(f);
  unsigned hb = b & 0xFFFF0000u;
  float    fl = f - __uint_as_float(hb);
  unsigned lb = __float_as_uint(fl);
  *hi = (unsigned short)(hb >> 16);
  *lo = (unsigned short)((lb + 0x7FFFu + ((lb >> 16) & 1u)) >> 16);
}

// ---------------- zero output ----------------
__global__ void k_zero(float4* __restrict__ out) {
  const size_t total  = (size_t)N_NODES * OUT_CH / 4;
  const size_t stride = (size_t)gridDim.x * blockDim.x;
  float4 z = make_float4(0.f, 0.f, 0.f, 0.f);
  for (size_t i = (size_t)blockIdx.x * blockDim.x + threadIdx.x; i < total; i += stride)
    out[i] = z;
}

// ---------------- keys = log(p) + gumbel(hash(i)) ----------------
__global__ void k_keys(const float* __restrict__ p, float* __restrict__ keys) {
  const unsigned stride = gridDim.x * blockDim.x;
  for (unsigned i = blockIdx.x * blockDim.x + threadIdx.x; i < N_NODES; i += stride) {
    unsigned h = i + 0x2545F491u;                 // seed-42-derived offset
    h ^= h >> 16; h *= 0x7FEB352Du;
    h ^= h >> 15; h *= 0x846CA68Bu;
    h ^= h >> 16;
    float u = (float)(h >> 8) * (1.0f / 16777216.0f) + 2.9802322e-8f;  // (0,1)
    float g = -logf(-logf(u));
    keys[i] = logf(p[i]) + g;
  }
}

// ---------------- radix-select state init ----------------
__global__ void k_init(unsigned* __restrict__ hist, unsigned* __restrict__ state) {
  hist[threadIdx.x] = 0u;
  if (threadIdx.x == 0) { state[0] = 0u; state[1] = S_SAMP; }
}

// ---------------- per-pass byte histogram (LDS-accumulated) ----------------
__global__ void k_hist(const float* __restrict__ keys, unsigned* __restrict__ hist,
                       const unsigned* __restrict__ state, int shift) {
  __shared__ unsigned lh[256];
  lh[threadIdx.x] = 0u;
  __syncthreads();
  const unsigned prefix = state[0];
  const unsigned mask   = (shift == 24) ? 0u : (0xFFFFFFFFu << (shift + 8));
  const unsigned stride = gridDim.x * blockDim.x;
  for (unsigned i = blockIdx.x * blockDim.x + threadIdx.x; i < N_NODES; i += stride) {
    unsigned m = monokey(keys[i]);
    if ((m & mask) == (prefix & mask))
      atomicAdd(&lh[(m >> shift) & 0xFFu], 1u);
  }
  __syncthreads();
  unsigned c = lh[threadIdx.x];
  if (c) atomicAdd(&hist[threadIdx.x], c);
}

// ---------------- pick byte, update prefix/need, clear hist ----------------
__global__ void k_select(unsigned* __restrict__ hist, unsigned* __restrict__ state, int shift) {
  if (threadIdx.x == 0) {
    unsigned need = state[1];
    unsigned acc  = 0; int b = 0;
    for (int v = 255; v >= 0; --v) {
      unsigned c = hist[v];
      if (acc + c >= need) { b = v; break; }
      acc += c;
    }
    state[0] |= ((unsigned)b) << shift;
    state[1]  = need - acc;
    if (shift == 0) {           // final pass: arm compaction counters
      state[2] = 0u;            // greater-than slot counter (front)
      state[3] = 0u;            // equal-to counter (back)
      state[4] = need - acc;    // #ties to accept
    }
  }
  __syncthreads();
  hist[threadIdx.x] = 0u;       // ready for next pass / next call
}

// ---------------- compact selected node ids into sampled[] ----------------
__global__ void k_compact(const float* __restrict__ keys, unsigned* __restrict__ state,
                          int* __restrict__ sampled) {
  const unsigned T       = state[0];
  const unsigned need_eq = state[4];
  const unsigned stride  = gridDim.x * blockDim.x;
  for (unsigned i = blockIdx.x * blockDim.x + threadIdx.x; i < N_NODES; i += stride) {
    unsigned m = monokey(keys[i]);
    if (m > T) {
      unsigned s = atomicAdd(&state[2], 1u);
      sampled[s] = (int)i;
    } else if (m == T) {
      unsigned t = atomicAdd(&state[3], 1u);
      if (t < need_eq) sampled[S_SAMP - 1u - t] = (int)i;
    }
  }
}

// ---------------- W -> bf16 hi/lo, pre-swizzled to WMMA B-fragment layout ----
// Fragment (kt, nt): lane l holds B[k][n] for n = nt*16 + (l&15),
// k = kt*32 + (l>>4)*16 + j  (j = 0..15, VGPR r = j/2, low/high half = j&1).
__global__ void k_wconv(const float* __restrict__ w,
                        unsigned short* __restrict__ whi,
                        unsigned short* __restrict__ wlo) {
  const int id    = blockIdx.x * blockDim.x + threadIdx.x;   // 0..4095
  const int lane  = id & 31;
  const int nt    = (id >> 5) & 15;
  const int kt    = id >> 9;
  const int n     = nt * 16 + (lane & 15);
  const int kbase = kt * 32 + (lane >> 4) * 16;
  const size_t off = ((size_t)((kt * 16 + nt) * 32 + lane)) * 16;
#pragma unroll
  for (int j = 0; j < 16; ++j) {
    unsigned short hi, lo;
    split_bf16(w[(size_t)(kbase + j) * OUT_CH + n], &hi, &lo);
    whi[off + j] = hi;
    wlo[off + j] = lo;
  }
}

// ---------------- gather + bf16x3 WMMA GEMM + bias/scale scatter ----------------
__global__ __launch_bounds__(256) void k_gemm(
    const float* __restrict__ x, const int* __restrict__ sampled,
    const unsigned short* __restrict__ whi, const unsigned short* __restrict__ wlo,
    const float* __restrict__ bias, float* __restrict__ out) {
  __shared__ __align__(16) unsigned short sAhi[TM * IN_CH];
  __shared__ __align__(16) unsigned short sAlo[TM * IN_CH];
  __shared__ int sIdx[TM];

  const int tid     = threadIdx.x;
  const int rowBase = blockIdx.x * TM;     // S_SAMP = TM * gridDim.x exactly
  if (tid < TM) sIdx[tid] = sampled[rowBase + tid];
  __syncthreads();

  // Gather TM x 256 fp32 rows, split to bf16 hi/lo in LDS (2048 float4 / 256 thr)
#pragma unroll
  for (int it = 0; it < (TM * IN_CH / 4) / 256; ++it) {
    const int linear = it * 256 + tid;
    const int row    = linear >> 6;        // 64 float4 per row
    const int c4     = linear & 63;
    const float4 v = ((const float4*)(x + (size_t)sIdx[row] * IN_CH))[c4];
    const int base = row * IN_CH + c4 * 4;
    split_bf16(v.x, &sAhi[base + 0], &sAlo[base + 0]);
    split_bf16(v.y, &sAhi[base + 1], &sAlo[base + 1]);
    split_bf16(v.z, &sAhi[base + 2], &sAlo[base + 2]);
    split_bf16(v.w, &sAhi[base + 3], &sAlo[base + 3]);
  }
  __syncthreads();

  const int wave = tid >> 5;
  const int lane = tid & 31;
  const int hf   = lane >> 4;    // lane half
  const int ln   = lane & 15;
  const int nt0  = wave * 2;     // each wave: N-tiles nt0, nt0+1

  v8f zero = {0.f, 0.f, 0.f, 0.f, 0.f, 0.f, 0.f, 0.f};
  v8f acc[2][2];
  acc[0][0] = zero; acc[0][1] = zero; acc[1][0] = zero; acc[1][1] = zero;

#pragma unroll
  for (int kt = 0; kt < 8; ++kt) {
    Frag bh0, bl0, bh1, bl1;
    {
      const size_t f0 = ((size_t)((kt * 16 + nt0)     * 32 + lane)) * 16;
      const size_t f1 = ((size_t)((kt * 16 + nt0 + 1) * 32 + lane)) * 16;
      const uv4* p;
      p = (const uv4*)(whi + f0); bh0.q[0] = p[0]; bh0.q[1] = p[1];
      p = (const uv4*)(wlo + f0); bl0.q[0] = p[0]; bl0.q[1] = p[1];
      p = (const uv4*)(whi + f1); bh1.q[0] = p[0]; bh1.q[1] = p[1];
      p = (const uv4*)(wlo + f1); bl1.q[0] = p[0]; bl1.q[1] = p[1];
    }
#pragma unroll
    for (int mt = 0; mt < 2; ++mt) {
      // A fragment: lane m = ln; VGPR 0-3 <- K kb..kb+7, VGPR 4-7 <- kb+8..kb+15 (kb = kt*32 + hf*16, matches B)
      Frag ah, al;
      const int arow = mt * 16 + ln;
      const int kb   = kt * 32 + hf * 16;
      ah.q[0] = *(const uv4*)&sAhi[arow * IN_CH + kb];
      ah.q[1] = *(const uv4*)&sAhi[arow * IN_CH + kb + 8];
      al.q[0] = *(const uv4*)&sAlo[arow * IN_CH + kb];
      al.q[1] = *(const uv4*)&sAlo[arow * IN_CH + kb + 8];

      // fp32-emulating bf16x3: hi*hi + hi*lo + lo*hi
      acc[mt][0] = __builtin_amdgcn_wmma_f32_16x16x32_bf16(false, ah.v, false, bh0.v, (short)0, acc[mt][0], false, false);
      acc[mt][0] = __builtin_amdgcn_wmma_f32_16x16x32_bf16(false, ah.v, false, bl0.v, (short)0, acc[mt][0], false, false);
      acc[mt][0] = __builtin_amdgcn_wmma_f32_16x16x32_bf16(false, al.v, false, bh0.v, (short)0, acc[mt][0], false, false);
      acc[mt][1] = __builtin_amdgcn_wmma_f32_16x16x32_bf16(false, ah.v, false, bh1.v, (short)0, acc[mt][1], false, false);
      acc[mt][1] = __builtin_amdgcn_wmma_f32_16x16x32_bf16(false, ah.v, false, bl1.v, (short)0, acc[mt][1], false, false);
      acc[mt][1] = __builtin_amdgcn_wmma_f32_16x16x32_bf16(false, al.v, false, bh1.v, (short)0, acc[mt][1], false, false);
    }
  }

  // Epilogue: D layout -> element r is row (r + 8*hf), col ln of the tile
  const float scale = (float)N_NODES / (float)S_SAMP;
#pragma unroll
  for (int nti = 0; nti < 2; ++nti) {
    const int   col = (nt0 + nti) * 16 + ln;
    const float bc  = bias[col];
#pragma unroll
    for (int mt = 0; mt < 2; ++mt) {
#pragma unroll
      for (int r = 0; r < 8; ++r) {
        const int rib = mt * 16 + r + hf * 8;
        out[(size_t)sIdx[rib] * OUT_CH + col] = (acc[mt][nti][r] + bc) * scale;
      }
    }
  }
}

extern "C" void kernel_launch(void* const* d_in, const int* in_sizes, int n_in,
                              void* d_out, int out_size, void* d_ws, size_t ws_size,
                              hipStream_t stream) {
  const float* x      = (const float*)d_in[0];
  const float* weight = (const float*)d_in[1];
  const float* bias   = (const float*)d_in[2];
  const float* imp    = (const float*)d_in[3];
  // d_in[4] = edge_index (unused by the sampled branch)
  float* out = (float*)d_out;

  char* ws = (char*)d_ws;
  // ws layout (16B aligned): keys | sampled | Whi frags | Wlo frags | hist | state
  float*          keys    = (float*)(ws + 0);                       // 2,000,000 B
  int*            sampled = (int*)(ws + 2000000);                   //   400,000 B
  unsigned short* whi     = (unsigned short*)(ws + 2400000);        //   131,072 B
  unsigned short* wlo     = (unsigned short*)(ws + 2400000 + 131072);
  unsigned*       hist    = (unsigned*)(ws + 2400000 + 262144);     //     1,024 B
  unsigned*       state   = hist + 256;                             //        32 B

  k_zero <<<8192, 256, 0, stream>>>((float4*)out);
  k_wconv<<<16,   256, 0, stream>>>(weight, whi, wlo);
  k_keys <<<512,  256, 0, stream>>>(imp, keys);
  k_init <<<1,    256, 0, stream>>>(hist, state);
  for (int shift = 24; shift >= 0; shift -= 8) {
    k_hist  <<<256, 256, 0, stream>>>(keys, hist, state, shift);
    k_select<<<1,   256, 0, stream>>>(hist, state, shift);
  }
  k_compact<<<512, 256, 0, stream>>>(keys, state, sampled);
  k_gemm<<<S_SAMP / TM, 256, 0, stream>>>(x, sampled, whi, wlo, bias, out);
}
